// Model_26104811225248
// MI455X (gfx1250) — matrix operand order repaired and checked by reading, not compiled, not run on hardware
//
#include <hip/hip_runtime.h>
#include <hip/hip_bf16.h>

#define B_   64
#define S_   512
#define D_   512
#define TI   64            // i-tile rows per block
#define TJC  128           // j-chunk columns per pass
#define TK   128           // K tile
#define PAD  8
#define LROW (TK + PAD)    // 136 halves per LDS row (272B stride, 16B aligned)
#define NEGF (-1e9f)

#define BUF_HALVES ((TI + TJC) * LROW)      // halves per ping-pong buffer
#define BOFF_HALVES (TI * LROW)             // B region offset within buffer

typedef _Float16 v16h  __attribute__((ext_vector_type(16)));
typedef _Float16 half8 __attribute__((ext_vector_type(8)));
typedef float    v8f   __attribute__((ext_vector_type(8)));

// CAS-based float atomic max: works on both LDS and global pointers, any sign.
__device__ inline void atomicMaxF(float* addr, float val) {
    unsigned* ua = (unsigned*)addr;
    float cur = *addr;
    while (val > cur) {
        unsigned assumed = __float_as_uint(cur);
        unsigned prev = atomicCAS(ua, assumed, __float_as_uint(val));
        if (prev == assumed) break;
        cur = __uint_as_float(prev);
    }
}

// CDNA5 async memory->LDS copy, 16 bytes per lane. GVS mode: 64-bit SGPR base +
// per-lane 32-bit byte offset. Tracked by ASYNCcnt (not LOADcnt) — no VGPR
// round-trip, no per-chunk s_wait_loadcnt.
__device__ __forceinline__ void async_ld16(unsigned lds_off, const void* sbase, unsigned voff) {
    asm volatile("global_load_async_to_lds_b128 %0, %1, %2"
                 :: "v"(lds_off), "v"(voff), "s"(sbase)
                 : "memory");
}
__device__ __forceinline__ void wait_async0() {
    asm volatile("s_wait_asynccnt 0" ::: "memory");
}

// ---------------------------------------------------------------------------
// 1) L2-normalize rows of [nrows, 512] fp32 -> f16. One wave32 per row.
// ---------------------------------------------------------------------------
__global__ __launch_bounds__(256)
void normalize_kernel(const float* __restrict__ in, _Float16* __restrict__ out, int nrows) {
    const int row  = blockIdx.x * 8 + (threadIdx.x >> 5);
    const int lane = threadIdx.x & 31;
    if (row >= nrows) return;

    const float* p = in + (size_t)row * D_ + lane * 16;
    float4 v0 = *(const float4*)(p + 0);
    float4 v1 = *(const float4*)(p + 4);
    float4 v2 = *(const float4*)(p + 8);
    float4 v3 = *(const float4*)(p + 12);

    float x[16] = {v0.x, v0.y, v0.z, v0.w, v1.x, v1.y, v1.z, v1.w,
                   v2.x, v2.y, v2.z, v2.w, v3.x, v3.y, v3.z, v3.w};
    float s = 0.f;
#pragma unroll
    for (int e = 0; e < 16; ++e) s = fmaf(x[e], x[e], s);
    s += __shfl_xor(s, 1);
    s += __shfl_xor(s, 2);
    s += __shfl_xor(s, 4);
    s += __shfl_xor(s, 8);
    s += __shfl_xor(s, 16);
    const float scale = 1.0f / fmaxf(sqrtf(s), 1e-8f);

    half8 h0, h1;
#pragma unroll
    for (int e = 0; e < 8; ++e) h0[e] = (_Float16)(x[e] * scale);
#pragma unroll
    for (int e = 0; e < 8; ++e) h1[e] = (_Float16)(x[8 + e] * scale);
    _Float16* q = out + (size_t)row * D_ + lane * 16;
    *(half8*)(q + 0) = h0;
    *(half8*)(q + 8) = h1;
}

// ---------------------------------------------------------------------------
// 2) init workspace reduction buffers
// ---------------------------------------------------------------------------
__global__ __launch_bounds__(256)
void init_kernel(float* __restrict__ colmax_g, float* __restrict__ rowsum_g) {
    const int i = blockIdx.x * 256 + threadIdx.x;
    if (i < B_ * S_) colmax_g[i] = NEGF;
    if (i < B_)      rowsum_g[i] = 0.f;
}

// ---------------------------------------------------------------------------
// 3) Batched f16 GEMM (e1n @ e2n^T) with fused masked row/col max reductions.
//    Grid: 64 batches x 8 i-tiles. Block: 256 threads = 8 wave32.
//    Wave w owns a 2x2 arrangement of 16x16 tiles:
//      i sub-tiles {2*(w&1), 2*(w&1)+1}, j sub-tiles {2*(w>>1), 2*(w>>1)+1}.
//    K is streamed through double-buffered LDS via async-to-LDS DMA.
// ---------------------------------------------------------------------------
__global__ __launch_bounds__(256)
void sim_kernel(const _Float16* __restrict__ e1n, const _Float16* __restrict__ e2n,
                const int* __restrict__ m1, const int* __restrict__ m2,
                float* __restrict__ colmax_g, float* __restrict__ rowsum_g) {
    __shared__ __align__(16) _Float16 Stage[2][BUF_HALVES];   // A(64 rows)+B(128 rows), x2
    __shared__ float m1f[TI];
    __shared__ float m2f[S_];
    __shared__ float rowmax[TI];
    __shared__ float colmax[S_];

    const int tid = threadIdx.x;
    const int b   = blockIdx.x >> 3;
    const int i0  = (blockIdx.x & 7) * TI;

    for (int j = tid; j < S_; j += 256) {
        m2f[j]    = (float)m2[b * S_ + j];
        colmax[j] = NEGF;
    }
    if (tid < TI) {
        m1f[tid]    = (float)m1[b * S_ + i0 + tid];
        rowmax[tid] = NEGF;
    }
    __syncthreads();

    const int lane = tid & 31;
    const int wave = tid >> 5;
    const int hf   = lane >> 4;        // half-wave select
    const int n16  = lane & 15;
    const int iw   = wave & 1;         // i-pair select (2 tiles)
    const int jw   = wave >> 1;        // j-pair select (2 tiles)

    // --- per-thread async staging offsets (A: 4 chunks, B: 8 chunks of 16B) ---
    const unsigned ldsBase = (unsigned)(uintptr_t)(&Stage[0][0]);
    unsigned voffA[4], ldsA[4];
#pragma unroll
    for (int q = 0; q < 4; ++q) {
        const int idx = tid + q * 256;
        const int row = idx >> 4;                 // 16 chunks of 8 halves per row
        const int ch  = (idx & 15) * 8;
        voffA[q] = (unsigned)(row * D_ + ch) * 2u;
        ldsA[q]  = ldsBase + (unsigned)(row * LROW + ch) * 2u;
    }
    unsigned voffB[8], ldsB[8];
#pragma unroll
    for (int q = 0; q < 8; ++q) {
        const int idx = tid + q * 256;
        const int row = idx >> 4;                 // 0..127
        const int ch  = (idx & 15) * 8;
        voffB[q] = (unsigned)(row * D_ + ch) * 2u;
        ldsB[q]  = ldsBase + (unsigned)(BOFF_HALVES + row * LROW + ch) * 2u;
    }

    const size_t baseA = ((size_t)b * S_ + i0) * D_;

    for (int jc = 0; jc < S_ / TJC; ++jc) {
        const int j0c = jc * TJC;
        const size_t baseB = ((size_t)b * S_ + j0c) * D_;
        v8f acc[2][2] = {};

        // prime buffer 0 with k0 = 0
        {
            const void* gA = (const void*)(e1n + baseA);
            const void* gB = (const void*)(e2n + baseB);
#pragma unroll
            for (int q = 0; q < 4; ++q) async_ld16(ldsA[q], gA, voffA[q]);
#pragma unroll
            for (int q = 0; q < 8; ++q) async_ld16(ldsB[q], gB, voffB[q]);
        }

        for (int t = 0; t < D_ / TK; ++t) {
            wait_async0();                       // my DMA into buf[t&1] done
            __syncthreads();                     // everyone's DMA done
            if (t + 1 < D_ / TK) {               // overlap next DMA with compute
                const unsigned bufb = (unsigned)(((t + 1) & 1) * BUF_HALVES * 2);
                const void* gA = (const void*)(e1n + baseA + (t + 1) * TK);
                const void* gB = (const void*)(e2n + baseB + (t + 1) * TK);
#pragma unroll
                for (int q = 0; q < 4; ++q) async_ld16(ldsA[q] + bufb, gA, voffA[q]);
#pragma unroll
                for (int q = 0; q < 8; ++q) async_ld16(ldsB[q] + bufb, gB, voffB[q]);
            }

            const _Float16* Ab = &Stage[t & 1][0];
            const _Float16* Bb = &Stage[t & 1][BOFF_HALVES];
#pragma unroll
            for (int kk = 0; kk < TK; kk += 32) {
                v16h a[2], bb[2];
#pragma unroll
                for (int ii = 0; ii < 2; ++ii) {
                    const int arow = (iw * 2 + ii) * 16 + n16;
                    half8 lo = *(const half8*)(Ab + arow * LROW + kk + 8 * hf);
                    half8 hi = *(const half8*)(Ab + arow * LROW + kk + 16 + 8 * hf);
                    a[ii] = __builtin_shufflevector(lo, hi,
                            0,1,2,3,4,5,6,7,8,9,10,11,12,13,14,15);
                }
#pragma unroll
                for (int jj = 0; jj < 2; ++jj) {
                    const int brow = (jw * 2 + jj) * 16 + n16;
                    half8 lo = *(const half8*)(Bb + brow * LROW + kk + 8 * hf);
                    half8 hi = *(const half8*)(Bb + brow * LROW + kk + 16 + 8 * hf);
                    bb[jj] = __builtin_shufflevector(lo, hi,
                            0,1,2,3,4,5,6,7,8,9,10,11,12,13,14,15);
                }
#pragma unroll
                for (int ii = 0; ii < 2; ++ii)
#pragma unroll
                    for (int jj = 0; jj < 2; ++jj)
                        acc[ii][jj] = __builtin_amdgcn_wmma_f32_16x16x32_f16(
                                false, a[ii], false, bb[jj], (short)0,
                                acc[ii][jj], false, false);
            }
        }

        // fused masked reductions for the four 16x16 tiles
#pragma unroll
        for (int ii = 0; ii < 2; ++ii) {
#pragma unroll
            for (int jj = 0; jj < 2; ++jj) {
                const v8f c = acc[ii][jj];
                const int jloc = j0c + (jw * 2 + jj) * 16 + n16;   // 0..511
                const float m2v = m2f[jloc];
                float cv = NEGF;
#pragma unroll
                for (int r = 0; r < 8; ++r) {
                    const int iloc = (iw * 2 + ii) * 16 + hf * 8 + r;  // 0..63
                    const float d = c[r];
                    cv = fmaxf(cv, (m1f[iloc] > 0.5f) ? d : NEGF);
                    float rv = (m2v > 0.5f) ? d : NEGF;
                    rv = fmaxf(rv, __shfl_xor(rv, 1));
                    rv = fmaxf(rv, __shfl_xor(rv, 2));
                    rv = fmaxf(rv, __shfl_xor(rv, 4));
                    rv = fmaxf(rv, __shfl_xor(rv, 8));
                    if (n16 == 0) atomicMaxF(&rowmax[iloc], rv);
                }
                cv = fmaxf(cv, __shfl_xor(cv, 16));
                if (hf == 0) atomicMaxF(&colmax[jloc], cv);
            }
        }
    }
    __syncthreads();

    // block epilogue: masked row-max sum -> atomicAdd; col maxes -> global atomic max
    if (tid < 32) {
        float rp = 0.f;
        for (int r = tid; r < TI; r += 32)
            rp += (m1f[r] > 0.5f) ? rowmax[r] : 0.f;
        rp += __shfl_xor(rp, 1);
        rp += __shfl_xor(rp, 2);
        rp += __shfl_xor(rp, 4);
        rp += __shfl_xor(rp, 8);
        rp += __shfl_xor(rp, 16);
        if (tid == 0) atomicAdd(&rowsum_g[b], rp);
    }
    for (int j = tid; j < S_; j += 256)
        atomicMaxF(&colmax_g[(size_t)b * S_ + j], colmax[j]);
}

// ---------------------------------------------------------------------------
// 4) finalize: out[b] = (row_sum + col_sum) / (n1 + n2)
// ---------------------------------------------------------------------------
__global__ __launch_bounds__(256)
void finalize_kernel(const int* __restrict__ m1, const int* __restrict__ m2,
                     const float* __restrict__ colmax_g,
                     const float* __restrict__ rowsum_g, float* __restrict__ out) {
    __shared__ float sC[256];
    __shared__ int   sN[256];
    const int b = blockIdx.x, t = threadIdx.x;
    int   n12 = 0;
    float cs  = 0.f;
    for (int j = t; j < S_; j += 256) {
        const int v1 = m1[b * S_ + j];
        const int v2 = m2[b * S_ + j];
        n12 += v1 + v2;
        if (v2) cs += colmax_g[(size_t)b * S_ + j];
    }
    sC[t] = cs; sN[t] = n12;
    __syncthreads();
    for (int s = 128; s > 0; s >>= 1) {
        if (t < s) { sC[t] += sC[t + s]; sN[t] += sN[t + s]; }
        __syncthreads();
    }
    if (t == 0) out[b] = (rowsum_g[b] + sC[0]) / (float)sN[0];
}

// ---------------------------------------------------------------------------
extern "C" void kernel_launch(void* const* d_in, const int* in_sizes, int n_in,
                              void* d_out, int out_size, void* d_ws, size_t ws_size,
                              hipStream_t stream) {
    (void)in_sizes; (void)n_in; (void)out_size; (void)ws_size;
    const float* emb1 = (const float*)d_in[0];
    const float* emb2 = (const float*)d_in[1];
    const int*   m1   = (const int*)d_in[2];
    const int*   m2   = (const int*)d_in[3];
    float* out = (float*)d_out;

    char* ws = (char*)d_ws;
    const size_t tensorBytes = (size_t)B_ * S_ * D_ * sizeof(_Float16);  // 33.5 MB
    _Float16* e1n      = (_Float16*)ws;
    _Float16* e2n      = (_Float16*)(ws + tensorBytes);
    float*    colmax_g = (float*)(ws + 2 * tensorBytes);
    float*    rowsum_g = colmax_g + B_ * S_;

    init_kernel<<<(B_ * S_ + 255) / 256, 256, 0, stream>>>(colmax_g, rowsum_g);
    normalize_kernel<<<(B_ * S_) / 8, 256, 0, stream>>>(emb1, e1n, B_ * S_);
    normalize_kernel<<<(B_ * S_) / 8, 256, 0, stream>>>(emb2, e2n, B_ * S_);
    sim_kernel<<<B_ * 8, 256, 0, stream>>>(e1n, e2n, m1, m2, colmax_g, rowsum_g);
    finalize_kernel<<<B_, 256, 0, stream>>>(m1, m2, colmax_g, rowsum_g, out);
}
